// LightweightTransformer_59931973648642
// MI455X (gfx1250) — matrix operand order repaired and checked by reading, not compiled, not run on hardware
//
#include <hip/hip_runtime.h>
#include <hip/hip_bf16.h>

// ---------------------------------------------------------------------------
// Types for CDNA5 WMMA (wave32): A/B = 16 x bf16 per lane, C/D = 8 x f32.
// ---------------------------------------------------------------------------
typedef __attribute__((ext_vector_type(16))) __bf16 v16bf;
typedef __attribute__((ext_vector_type(8)))  float  v8f;
typedef __attribute__((ext_vector_type(8)))  unsigned short ushort8;

// Fragment built from two contiguous 16-byte loads -> ds_load_b128 pairs.
union BF16Frag {
    ushort8 h[2];
    v16bf   v;
};

__device__ __forceinline__ unsigned short f2bf_u(float f) {
    unsigned int u = __builtin_bit_cast(unsigned int, f);
    u += 0x7FFFu + ((u >> 16) & 1u);          // round-to-nearest-even
    return (unsigned short)(u >> 16);
}

// ---------------------------------------------------------------------------
// Tiled transpose + fp32->bf16 convert:  dst[N,K] = bf16(src[K,N])
// blockIdx.z = layer.  Coalesced reads and writes via 32x33 LDS tile.
// ---------------------------------------------------------------------------
__global__ __launch_bounds__(256)
void transp_f32_to_bf16(const float* __restrict__ src,
                        unsigned short* __restrict__ dst, int K, int N) {
    __shared__ unsigned short tile[32][33];
    const int n0 = blockIdx.x * 32, k0 = blockIdx.y * 32;
    const int tx = threadIdx.x & 31, ty = threadIdx.x >> 5;   // 32 x 8
    src += (size_t)blockIdx.z * K * N;
    dst += (size_t)blockIdx.z * K * N;
#pragma unroll
    for (int i = 0; i < 32; i += 8)
        tile[ty + i][tx] = f2bf_u(src[(size_t)(k0 + ty + i) * N + n0 + tx]);
    __syncthreads();
#pragma unroll
    for (int i = 0; i < 32; i += 8)
        dst[(size_t)(n0 + ty + i) * K + k0 + tx] = tile[tx][ty + i];
}

// ---------------------------------------------------------------------------
// bf16 1024x1024 transpose per batch:  dst[b,d,s] = src[b,s,d]
// Produces V^T so flash-attention P.V fragments are LDS-contiguous.
// ---------------------------------------------------------------------------
__global__ __launch_bounds__(256)
void transp_bf16(const unsigned short* __restrict__ src,
                 unsigned short* __restrict__ dst) {
    __shared__ unsigned short tile[32][33];
    const int d0 = blockIdx.x * 32, s0 = blockIdx.y * 32;
    const int tx = threadIdx.x & 31, ty = threadIdx.x >> 5;
    const size_t base = (size_t)blockIdx.z * 1024 * 1024;
#pragma unroll
    for (int i = 0; i < 32; i += 8)
        tile[ty + i][tx] = src[base + (size_t)(s0 + ty + i) * 1024 + d0 + tx];
    __syncthreads();
#pragma unroll
    for (int i = 0; i < 32; i += 8)
        dst[base + (size_t)(d0 + ty + i) * 1024 + s0 + tx] = tile[tx][ty + i];
}

// ---------------------------------------------------------------------------
// Embedding: x[b,s,:] = emb[ids[b,s],:] + pos_emb[s,:]  (fp32 + bf16 copies)
// ---------------------------------------------------------------------------
__global__ __launch_bounds__(256)
void embed_k(const int* __restrict__ ids, const float* __restrict__ emb,
             const float* __restrict__ pos, float* __restrict__ xf,
             unsigned short* __restrict__ xb) {
    const int row = blockIdx.x;          // b*S + s
    const int s = row & 1023;
    const int t = threadIdx.x;
    const int id = ids[row];
#pragma unroll
    for (int i = 0; i < 4; ++i) {
        int d = t + i * 256;
        float v = emb[(size_t)id * 1024 + d] + pos[(size_t)s * 1024 + d];
        xf[(size_t)row * 1024 + d] = v;
        xb[(size_t)row * 1024 + d] = f2bf_u(v);
    }
}

// ---------------------------------------------------------------------------
// WMMA GEMM: C[M,N] = A[M,K](bf16) x BT[N,K](bf16, pre-transposed) + bias.
// Block = 128 threads (4 waves). Tile 128x64, K-step 32, double-buffered LDS:
// next tile is loaded to registers BEFORE the 8 WMMAs of the current tile, and
// stored to the alternate buffer after -> global latency hidden by matrix math,
// one barrier per K-step.  All fragments are contiguous 2x16B LDS loads.
// ---------------------------------------------------------------------------
__global__ __launch_bounds__(128)
void gemm_bf16_wmma(const unsigned short* __restrict__ A,
                    const unsigned short* __restrict__ BT,
                    const float* __restrict__ bias,
                    float* __restrict__ Cf, unsigned short* __restrict__ Cb,
                    int M, int N, int K, int act) {
    __shared__ unsigned short As[2][128][32];
    __shared__ unsigned short Bs[2][64][32];
    const int t    = threadIdx.x;
    const int lane = t & 31;
    const int w    = t >> 5;
    const int n0   = blockIdx.x * 64;
    const int m0   = blockIdx.y * 128;

    const int rlo = lane & 15;
    const int kb  = (lane < 16) ? 0 : 8;    // A/P fragment K-base
    const int kb2 = (lane < 16) ? 0 : 16;   // unused; B fragment also uses kb
    const int hi  = (lane < 16) ? 0 : 8;

    v8f acc[2][4] = {};

    const int brow = t >> 1, bch = (t & 1) * 16;
    const unsigned short* apBase = A  + (size_t)(m0 + t) * K;
    const unsigned short* bpBase = BT + (size_t)(n0 + brow) * K + bch;

    // Prologue: stage K-slab 0 into buffer 0
    ushort8 ra[4], rb[2];
#pragma unroll
    for (int i = 0; i < 4; ++i) ra[i] = *(const ushort8*)(apBase + i * 8);
    rb[0] = *(const ushort8*)(bpBase);
    rb[1] = *(const ushort8*)(bpBase + 8);
#pragma unroll
    for (int i = 0; i < 4; ++i) *(ushort8*)&As[0][t][i * 8] = ra[i];
    *(ushort8*)&Bs[0][brow][bch]     = rb[0];
    *(ushort8*)&Bs[0][brow][bch + 8] = rb[1];
    __syncthreads();

    int buf = 0;
    for (int k0 = 0; k0 < K; k0 += 32, buf ^= 1) {
        const bool hasNext = (k0 + 32) < K;
        if (hasNext) {      // issue next-slab global loads before computing
            const unsigned short* ap = apBase + k0 + 32;
            const unsigned short* bp = bpBase + k0 + 32;
#pragma unroll
            for (int i = 0; i < 4; ++i) ra[i] = *(const ushort8*)(ap + i * 8);
            rb[0] = *(const ushort8*)(bp);
            rb[1] = *(const ushort8*)(bp + 8);
            if (k0 + 64 < K) {              // -> global_prefetch_b8
                __builtin_prefetch(ap + 32, 0, 3);
                __builtin_prefetch(bp + 32, 0, 3);
            }
        }

        // 2 A-frags x 4 B-frags -> 8 WMMA on current buffer
        BF16Frag af[2];
#pragma unroll
        for (int r = 0; r < 2; ++r) {
            const unsigned short* as = &As[buf][w * 32 + r * 16 + rlo][0];
            af[r].h[0] = *(const ushort8*)(as + kb);
            af[r].h[1] = *(const ushort8*)(as + 16 + kb);
        }
#pragma unroll
        for (int j = 0; j < 4; ++j) {
            BF16Frag bf;
            const unsigned short* bs = &Bs[buf][j * 16 + rlo][0];
            bf.h[0] = *(const ushort8*)(bs + kb);
            bf.h[1] = *(const ushort8*)(bs + 16 + kb);
#pragma unroll
            for (int r = 0; r < 2; ++r)
                acc[r][j] = __builtin_amdgcn_wmma_f32_16x16x32_bf16(
                    false, af[r].v, false, bf.v, (short)0, acc[r][j], false, false);
        }

        if (hasNext) {      // drain loads after WMMAs, fill alternate buffer
#pragma unroll
            for (int i = 0; i < 4; ++i) *(ushort8*)&As[buf ^ 1][t][i * 8] = ra[i];
            *(ushort8*)&Bs[buf ^ 1][brow][bch]     = rb[0];
            *(ushort8*)&Bs[buf ^ 1][brow][bch + 8] = rb[1];
            __syncthreads();
        }
    }

#pragma unroll
    for (int j = 0; j < 4; ++j) {
        const int n  = n0 + j * 16 + rlo;
        const float bn = bias[n];
#pragma unroll
        for (int r = 0; r < 2; ++r) {
#pragma unroll
            for (int v = 0; v < 8; ++v) {
                const int m = m0 + w * 32 + r * 16 + v + hi;
                float x = acc[r][j][v] + bn;
                if (act) x = 0.5f * x * (1.0f + erff(x * 0.70710678118654752f));
                size_t idx = (size_t)m * N + n;
                if (Cf) Cf[idx] = x;
                if (Cb) Cb[idx] = f2bf_u(x);
            }
        }
    }
}

// ---------------------------------------------------------------------------
// Flash attention: one wave per (b, h, 16-row q tile). Online softmax over
// 32-key chunks. QK^T and P.V on v_wmma_f32_16x16x32_bf16.  K/V/mask for
// chunk c+1 are register-prefetched while chunk c computes.
// Q,K: [B,S,D] bf16 (head h at cols h*64..).  VT: [B, D, S] bf16 (V^T).
// ---------------------------------------------------------------------------
__global__ __launch_bounds__(32)
void flash_attn(const unsigned short* __restrict__ Q,
                const unsigned short* __restrict__ Kk,
                const unsigned short* __restrict__ VT,
                const int* __restrict__ amask,
                unsigned short* __restrict__ O) {
    __shared__ unsigned short Kt[32][64];   // [krow][d]
    __shared__ unsigned short Vt[64][32];   // [d][krow]  (from V^T: contiguous)
    __shared__ unsigned short Pt[16][32];
    __shared__ float Sraw[16][32];
    __shared__ float rowm[16], rowl[16], rowc[16];
    __shared__ int   msk[32];

    const int lane = threadIdx.x;
    const int qt = blockIdx.x & 63;
    const int h  = (blockIdx.x >> 6) & 15;
    const int b  = blockIdx.x >> 10;

    const int rlo = lane & 15;
    const int kb  = (lane < 16) ? 0 : 8;
    const int kb2 = (lane < 16) ? 0 : 16;
    const int hi  = (lane < 16) ? 0 : 8;

    // Q fragments (rows qt*16..+16, head dim in two K=32 slabs) - 4 x b128
    const unsigned short* qp =
        Q + ((size_t)(b * 1024 + qt * 16 + rlo)) * 1024 + h * 64;
    BF16Frag q0, q1;
    q0.h[0] = *(const ushort8*)(qp + kb);
    q0.h[1] = *(const ushort8*)(qp + 16 + kb);
    q1.h[0] = *(const ushort8*)(qp + 32 + kb);
    q1.h[1] = *(const ushort8*)(qp + 48 + kb);

    // Prologue: stage chunk 0 (K rows, V^T rows, mask)
    ushort8 rk[8], rv[8];
    int mreg;
    {
        const unsigned short* kp = Kk + ((size_t)(b * 1024 + lane)) * 1024 + h * 64;
#pragma unroll
        for (int j = 0; j < 8; ++j) rk[j] = *(const ushort8*)(kp + j * 8);
#pragma unroll
        for (int rr = 0; rr < 2; ++rr) {
            const unsigned short* vp =
                VT + ((size_t)(b * 1024 + h * 64 + lane + rr * 32)) * 1024;
#pragma unroll
            for (int j = 0; j < 4; ++j) rv[rr * 4 + j] = *(const ushort8*)(vp + j * 8);
        }
        mreg = amask[b * 1024 + lane];
    }
#pragma unroll
    for (int j = 0; j < 8; ++j) *(ushort8*)&Kt[lane][j * 8] = rk[j];
#pragma unroll
    for (int rr = 0; rr < 2; ++rr)
#pragma unroll
        for (int j = 0; j < 4; ++j)
            *(ushort8*)&Vt[lane + rr * 32][j * 8] = rv[rr * 4 + j];
    msk[lane] = mreg;
    if (lane < 16) { rowm[lane] = -3.0e38f; rowl[lane] = 0.f; }
    v8f acc[4] = {};
    __syncthreads();

    for (int c = 0; c < 32; ++c) {
        const bool hasNext = (c + 1) < 32;
        if (hasNext) {   // register-prefetch chunk c+1 while chunk c computes
            const int krow = (c + 1) * 32 + lane;
            const unsigned short* kp =
                Kk + ((size_t)(b * 1024 + krow)) * 1024 + h * 64;
#pragma unroll
            for (int j = 0; j < 8; ++j) rk[j] = *(const ushort8*)(kp + j * 8);
#pragma unroll
            for (int rr = 0; rr < 2; ++rr) {
                const unsigned short* vp =
                    VT + ((size_t)(b * 1024 + h * 64 + lane + rr * 32)) * 1024
                       + (c + 1) * 32;
#pragma unroll
                for (int j = 0; j < 4; ++j)
                    rv[rr * 4 + j] = *(const ushort8*)(vp + j * 8);
            }
            mreg = amask[b * 1024 + krow];
        }

        // S = Q K^T / 8, masked; 2 key-col tiles x (2 chained WMMA)
#pragma unroll
        for (int j = 0; j < 2; ++j) {
            const unsigned short* ks = &Kt[j * 16 + rlo][0];
            BF16Frag bk0, bk1;
            bk0.h[0] = *(const ushort8*)(ks + kb);
            bk0.h[1] = *(const ushort8*)(ks + 16 + kb);
            bk1.h[0] = *(const ushort8*)(ks + 32 + kb);
            bk1.h[1] = *(const ushort8*)(ks + 48 + kb);
            v8f sc = {};
            sc = __builtin_amdgcn_wmma_f32_16x16x32_bf16(
                false, q0.v, false, bk0.v, (short)0, sc, false, false);
            sc = __builtin_amdgcn_wmma_f32_16x16x32_bf16(
                false, q1.v, false, bk1.v, (short)0, sc, false, false);
            const int kc = j * 16 + rlo;
            const bool dead = (msk[kc] == 0);
#pragma unroll
            for (int v = 0; v < 8; ++v) {
                float s = sc[v] * 0.125f;
                if (dead) s = -1.0e9f;
                Sraw[v + hi][kc] = s;
            }
        }
        __syncthreads();

        // Online softmax row update (one lane per row)
        if (lane < 16) {
            float mprev = rowm[lane];
            float rmax  = mprev;
            for (int kc = 0; kc < 32; ++kc) rmax = fmaxf(rmax, Sraw[lane][kc]);
            float corr = __expf(mprev - rmax);
            float psum = 0.f;
            for (int kc = 0; kc < 32; ++kc) {
                float p = __expf(Sraw[lane][kc] - rmax);
                Pt[lane][kc] = f2bf_u(p);
                psum += p;
            }
            rowl[lane] = rowl[lane] * corr + psum;
            rowm[lane] = rmax;
            rowc[lane] = corr;
        }
        __syncthreads();

        // acc = acc * corr + P @ V
        BF16Frag ap;
        {
            const unsigned short* ps = &Pt[rlo][0];
            ap.h[0] = *(const ushort8*)(ps + kb);
            ap.h[1] = *(const ushort8*)(ps + 16 + kb);
        }
#pragma unroll
        for (int jd = 0; jd < 4; ++jd) {
#pragma unroll
            for (int v = 0; v < 8; ++v) acc[jd][v] *= rowc[v + hi];
            const unsigned short* vs = &Vt[jd * 16 + rlo][0];
            BF16Frag bv;
            bv.h[0] = *(const ushort8*)(vs + kb);
            bv.h[1] = *(const ushort8*)(vs + 16 + kb);
            acc[jd] = __builtin_amdgcn_wmma_f32_16x16x32_bf16(
                false, ap.v, false, bv.v, (short)0, acc[jd], false, false);
        }
        __syncthreads();                     // chunk-c LDS fully consumed

        if (hasNext) {                       // commit prefetched chunk c+1
#pragma unroll
            for (int j = 0; j < 8; ++j) *(ushort8*)&Kt[lane][j * 8] = rk[j];
#pragma unroll
            for (int rr = 0; rr < 2; ++rr)
#pragma unroll
                for (int j = 0; j < 4; ++j)
                    *(ushort8*)&Vt[lane + rr * 32][j * 8] = rv[rr * 4 + j];
            msk[lane] = mreg;
            __syncthreads();
        }
    }

    // ctx = acc / l  -> bf16 [B,S,D] head-interleaved
#pragma unroll
    for (int jd = 0; jd < 4; ++jd) {
#pragma unroll
        for (int v = 0; v < 8; ++v) {
            const int m = v + hi;
            float val = acc[jd][v] / rowl[m];
            O[((size_t)(b * 1024 + qt * 16 + m)) * 1024 + h * 64 + jd * 16 + rlo]
                = f2bf_u(val);
        }
    }
}

// ---------------------------------------------------------------------------
// x = LayerNorm(xin + add) * g + b ; writes fp32 (in-place ok) + bf16 copy
// ---------------------------------------------------------------------------
__global__ __launch_bounds__(256)
void add_layernorm(const float* xin, const float* __restrict__ add,
                   const float* __restrict__ g, const float* __restrict__ bta,
                   float* xout, unsigned short* __restrict__ xbf) {
    __shared__ float red[256];
    __shared__ float s_mu, s_rstd;
    const int row = blockIdx.x, t = threadIdx.x;
    const float* xi = xin + (size_t)row * 1024;
    const float* ad = add + (size_t)row * 1024;
    float v[4];
    float s = 0.f;
#pragma unroll
    for (int i = 0; i < 4; ++i) { v[i] = xi[t + i * 256] + ad[t + i * 256]; s += v[i]; }
    red[t] = s; __syncthreads();
    for (int o = 128; o > 0; o >>= 1) { if (t < o) red[t] += red[t + o]; __syncthreads(); }
    if (t == 0) s_mu = red[0] * (1.f / 1024.f);
    __syncthreads();
    const float mu = s_mu;
    float q = 0.f;
#pragma unroll
    for (int i = 0; i < 4; ++i) { float d = v[i] - mu; q += d * d; }
    red[t] = q; __syncthreads();
    for (int o = 128; o > 0; o >>= 1) { if (t < o) red[t] += red[t + o]; __syncthreads(); }
    if (t == 0) s_rstd = rsqrtf(red[0] * (1.f / 1024.f) + 1e-5f);
    __syncthreads();
    const float rs = s_rstd;
#pragma unroll
    for (int i = 0; i < 4; ++i) {
        int d = t + i * 256;
        float y = (v[i] - mu) * rs * g[d] + bta[d];
        xout[(size_t)row * 1024 + d] = y;
        xbf[(size_t)row * 1024 + d]  = f2bf_u(y);
    }
}

// ---------------------------------------------------------------------------
// Classifier head: logits[b,:] = LN(x[b,0,:]) @ Wc + bc   (1024 x 5, scalar)
// ---------------------------------------------------------------------------
__global__ __launch_bounds__(256)
void cls_head(const float* __restrict__ x, const float* __restrict__ g,
              const float* __restrict__ bta, const float* __restrict__ Wc,
              const float* __restrict__ bc, float* __restrict__ out) {
    __shared__ float red[256];
    __shared__ float s_mu, s_rstd;
    __shared__ float nrm[1024];
    const int b = blockIdx.x, t = threadIdx.x;
    const float* xi = x + (size_t)b * 1024 * 1024;   // token 0 of batch b
    float v[4];
    float s = 0.f;
#pragma unroll
    for (int i = 0; i < 4; ++i) { v[i] = xi[t + i * 256]; s += v[i]; }
    red[t] = s; __syncthreads();
    for (int o = 128; o > 0; o >>= 1) { if (t < o) red[t] += red[t + o]; __syncthreads(); }
    if (t == 0) s_mu = red[0] * (1.f / 1024.f);
    __syncthreads();
    const float mu = s_mu;
    float q = 0.f;
#pragma unroll
    for (int i = 0; i < 4; ++i) { float d = v[i] - mu; q += d * d; }
    red[t] = q; __syncthreads();
    for (int o = 128; o > 0; o >>= 1) { if (t < o) red[t] += red[t + o]; __syncthreads(); }
    if (t == 0) s_rstd = rsqrtf(red[0] * (1.f / 1024.f) + 1e-5f);
    __syncthreads();
    const float rs = s_rstd;
#pragma unroll
    for (int i = 0; i < 4; ++i) {
        int d = t + i * 256;
        nrm[d] = (v[i] - mu) * rs * g[d] + bta[d];
    }
    __syncthreads();
    for (int c = 0; c < 5; ++c) {
        float p = 0.f;
#pragma unroll
        for (int i = 0; i < 4; ++i) { int d = t + i * 256; p += nrm[d] * Wc[d * 5 + c]; }
        red[t] = p; __syncthreads();
        for (int o = 128; o > 0; o >>= 1) { if (t < o) red[t] += red[t + o]; __syncthreads(); }
        if (t == 0) out[b * 5 + c] = red[0] + bc[c];
        __syncthreads();
    }
}

// ---------------------------------------------------------------------------
// Launch sequence
// ---------------------------------------------------------------------------
extern "C" void kernel_launch(void* const* d_in, const int* in_sizes, int n_in,
                              void* d_out, int out_size, void* d_ws, size_t ws_size,
                              hipStream_t stream) {
    (void)in_sizes; (void)n_in; (void)out_size; (void)ws_size;

    const int*   ids   = (const int*)  d_in[0];
    const int*   amask = (const int*)  d_in[1];
    const float* emb   = (const float*)d_in[2];
    const float* pos   = (const float*)d_in[3];
    const float* Wq    = (const float*)d_in[4];
    const float* bq    = (const float*)d_in[5];
    const float* Wk    = (const float*)d_in[6];
    const float* bk    = (const float*)d_in[7];
    const float* Wv    = (const float*)d_in[8];
    const float* bv    = (const float*)d_in[9];
    const float* Wo    = (const float*)d_in[10];
    const float* bo    = (const float*)d_in[11];
    const float* ln1g  = (const float*)d_in[12];
    const float* ln1b  = (const float*)d_in[13];
    const float* ln2g  = (const float*)d_in[14];
    const float* ln2b  = (const float*)d_in[15];
    const float* W1    = (const float*)d_in[16];
    const float* b1    = (const float*)d_in[17];
    const float* W2    = (const float*)d_in[18];
    const float* b2    = (const float*)d_in[19];
    const float* clsg  = (const float*)d_in[20];
    const float* clsb  = (const float*)d_in[21];
    const float* Wc    = (const float*)d_in[22];
    const float* bc    = (const float*)d_in[23];
    float* out = (float*)d_out;

    const size_t DD   = (size_t)1024 * 1024;
    const size_t DF   = (size_t)1024 * 4096;
    const size_t TOKD = (size_t)8192 * 1024;
    const size_t TOKF = (size_t)8192 * 4096;

    char* ws = (char*)d_ws;
    size_t off = 0;
    auto carve = [&](size_t bytes) -> void* {
        void* p = ws + off;
        off = (off + bytes + 255) & ~(size_t)255;
        return p;
    };
    unsigned short* wqT  = (unsigned short*)carve(3 * DD * 2);  // [N,K] bf16
    unsigned short* wkT  = (unsigned short*)carve(3 * DD * 2);
    unsigned short* wvT  = (unsigned short*)carve(3 * DD * 2);
    unsigned short* woT  = (unsigned short*)carve(3 * DD * 2);
    unsigned short* w1T  = (unsigned short*)carve(3 * DF * 2);  // [4096,1024]
    unsigned short* w2T  = (unsigned short*)carve(3 * DF * 2);  // [1024,4096]
    float*          xF   = (float*)         carve(TOKD * 4);
    unsigned short* xB   = (unsigned short*)carve(TOKD * 2);
    unsigned short* qB   = (unsigned short*)carve(TOKD * 2);
    unsigned short* kB   = (unsigned short*)carve(TOKD * 2);
    unsigned short* vB   = (unsigned short*)carve(TOKD * 2);
    unsigned short* vtB  = (unsigned short*)carve(TOKD * 2);    // V^T per batch
    unsigned short* ctxB = (unsigned short*)carve(TOKD * 2);
    float*          tmpF = (float*)         carve(TOKD * 4);
    unsigned short* f1B  = (unsigned short*)carve(TOKF * 2);

    // Weight convert + transpose (bf16 weights total 75 MB -> L2-resident)
    transp_f32_to_bf16<<<dim3(32, 32, 3),  256, 0, stream>>>(Wq, wqT, 1024, 1024);
    transp_f32_to_bf16<<<dim3(32, 32, 3),  256, 0, stream>>>(Wk, wkT, 1024, 1024);
    transp_f32_to_bf16<<<dim3(32, 32, 3),  256, 0, stream>>>(Wv, wvT, 1024, 1024);
    transp_f32_to_bf16<<<dim3(32, 32, 3),  256, 0, stream>>>(Wo, woT, 1024, 1024);
    transp_f32_to_bf16<<<dim3(128, 32, 3), 256, 0, stream>>>(W1, w1T, 1024, 4096);
    transp_f32_to_bf16<<<dim3(32, 128, 3), 256, 0, stream>>>(W2, w2T, 4096, 1024);

    embed_k<<<8192, 256, 0, stream>>>(ids, emb, pos, xF, xB);

    for (int l = 0; l < 3; ++l) {
        const size_t oDD = (size_t)l * DD, oDF = (size_t)l * DF;
        gemm_bf16_wmma<<<dim3(16, 64), 128, 0, stream>>>(
            xB, wqT + oDD, bq + l * 1024, nullptr, qB, 8192, 1024, 1024, 0);
        gemm_bf16_wmma<<<dim3(16, 64), 128, 0, stream>>>(
            xB, wkT + oDD, bk + l * 1024, nullptr, kB, 8192, 1024, 1024, 0);
        gemm_bf16_wmma<<<dim3(16, 64), 128, 0, stream>>>(
            xB, wvT + oDD, bv + l * 1024, nullptr, vB, 8192, 1024, 1024, 0);

        transp_bf16<<<dim3(32, 32, 8), 256, 0, stream>>>(vB, vtB);
        flash_attn<<<8192, 32, 0, stream>>>(qB, kB, vtB, amask, ctxB);

        gemm_bf16_wmma<<<dim3(16, 64), 128, 0, stream>>>(
            ctxB, woT + oDD, bo + l * 1024, tmpF, nullptr, 8192, 1024, 1024, 0);
        add_layernorm<<<8192, 256, 0, stream>>>(
            xF, tmpF, ln1g + l * 1024, ln1b + l * 1024, xF, xB);

        gemm_bf16_wmma<<<dim3(64, 64), 128, 0, stream>>>(
            xB, w1T + oDF, b1 + l * 4096, nullptr, f1B, 8192, 4096, 1024, 1);
        gemm_bf16_wmma<<<dim3(16, 64), 128, 0, stream>>>(
            f1B, w2T + oDF, b2 + l * 1024, tmpF, nullptr, 8192, 1024, 4096, 0);
        add_layernorm<<<8192, 256, 0, stream>>>(
            xF, tmpF, ln2g + l * 1024, ln2b + l * 1024, xF, xB);
    }

    cls_head<<<8, 256, 0, stream>>>(xF, clsg, clsb, Wc, bc, out);
}
